// MultiHeadAttention_82248623719138
// MI455X (gfx1250) — matrix operand order repaired and checked by reading, not compiled, not run on hardware
//
#include <hip/hip_runtime.h>
#include <hip/hip_bf16.h>

// MHA with per-head projections on MI455X (gfx1250, wave32, WMMA).
// Pipeline: repack weights f32->f16^T -> Q/K/V projection GEMMs (WMMA) ->
// flash attention (async-to-LDS double-buffered K/V staging, WMMA scores +
// online softmax + WMMA PV) -> output GEMM (WMMA).

typedef __attribute__((ext_vector_type(16))) _Float16 v16h;
typedef __attribute__((ext_vector_type(8)))  _Float16 v8h;
typedef __attribute__((ext_vector_type(8)))  float    v8f;

#define B_   2
#define H_   16
#define S_   2048
#define D_   128
#define DIM_ 2048
#define F_   256
#define SCALE_ 0.0625f   // 1/sqrt(F) = 1/16

// ---------------------------------------------------------------- WMMA core
__device__ __forceinline__ v8f wmma_fp16(v16h a, v16h b, v8f c) {
  // D = A(16x32 f16) * B(32x16 f16) + C(16x16 f32)
  return __builtin_amdgcn_wmma_f32_16x16x32_f16(
      /*neg_a=*/false, a, /*neg_b=*/false, b,
      /*c_mod=*/(short)0, c, /*reuse_a=*/false, /*reuse_b=*/false);
}

// ---------------------------------------------- CDNA5 async-to-LDS (inline asm)
// GLOBAL_LOAD_ASYNC_TO_LDS_B128: per-lane DMA of 16B global -> LDS, tracked by
// ASYNCcnt (CDNA5 ISA). vdst = LDS byte address VGPR.
// Generic pointers to __shared__ carry the LDS offset in their low 32 bits
// (aperture check uses addr[63:32] only).
__device__ __forceinline__ void async_ld_b128(void* lds_ptr, const void* gptr) {
  unsigned lds_off = (unsigned)(unsigned long long)lds_ptr;
  asm volatile("global_load_async_to_lds_b128 %0, %1, off"
               :: "v"(lds_off), "v"(gptr)
               : "memory");
}
__device__ __forceinline__ void wait_async0() {
  asm volatile("s_wait_asynccnt 0" ::: "memory");
}

// A-fragment (16x32, row-major source): lane m = lane&15,
// halves = row[kb..kb+7] ++ row[kb+16..kb+23], kb = 8*(lane>=16) + 32*chunk.
__device__ __forceinline__ v16h load_a_f16(const _Float16* __restrict__ row, int kb) {
  v8h lo = *(const v8h*)(row + kb);
  v8h hi = *(const v8h*)(row + kb + 16);
  v16h f;
#pragma unroll
  for (int i = 0; i < 8; ++i) { f[i] = lo[i]; f[8 + i] = hi[i]; }
  return f;
}

// Same A-fragment but converting f32 source on the fly (v_cvt_pk_f16_f32).
__device__ __forceinline__ v16h load_a_f32(const float* __restrict__ row, int kb) {
  v16h f;
#pragma unroll
  for (int i = 0; i < 8; ++i) {
    f[i]     = (_Float16)row[kb + i];
    f[8 + i] = (_Float16)row[kb + 16 + i];
  }
  return f;
}

// B-fragment (32x16) from a row of B^T (i.e. column n of B stored contiguously):
// halves = btrow[kb..kb+7] ++ btrow[kb+16..kb+23], kb = 8*(lane>=16) + 32*chunk.
__device__ __forceinline__ v16h load_b_frag(const _Float16* __restrict__ btrow, int kb) {
  v8h lo = *(const v8h*)(btrow + kb);
  v8h hi = *(const v8h*)(btrow + kb + 16);
  v16h f;
#pragma unroll
  for (int i = 0; i < 8; ++i) { f[i] = lo[i]; f[8 + i] = hi[i]; }
  return f;
}

// Row reductions across a 16-lane group (C-fragment rows live across 16 lanes).
__device__ __forceinline__ float rowmax16(float x) {
#pragma unroll
  for (int m = 1; m < 16; m <<= 1) x = fmaxf(x, __shfl_xor(x, m, 16));
  return x;
}
__device__ __forceinline__ float rowsum16(float x) {
#pragma unroll
  for (int m = 1; m < 16; m <<= 1) x += __shfl_xor(x, m, 16);
  return x;
}

// ------------------------------------------- K0: weight convert + transpose
// src: f32 [batch][R][C] -> dst: f16 [batch][C][R]
__global__ void cvt_transpose_kernel(const float* __restrict__ src,
                                     _Float16* __restrict__ dst,
                                     int R, int C, int batch) {
  long long total = (long long)batch * R * C;
  long long stride = (long long)gridDim.x * blockDim.x;
  for (long long i = (long long)blockIdx.x * blockDim.x + threadIdx.x;
       i < total; i += stride) {
    long long bidx = i / ((long long)R * C);
    long long rc   = i - bidx * (long long)R * C;
    int r = (int)(rc / C), c = (int)(rc % C);
    dst[bidx * (long long)R * C + (long long)c * R + r] = (_Float16)src[i];
  }
}

// --------------------------------- K1: per-head projection GEMM (WMMA, N=128)
// Y[b,h,s,:] = X[b,h,s,:] @ W[h] + bias[h];  WT is f16 [H][128][Fin] (= W^T)
// transpose_out=0 -> Y f16 [B,H,S,128];  transpose_out=1 -> Y f16 [B,H,128,S]
__global__ void proj_kernel(const float* __restrict__ X,
                            const _Float16* __restrict__ WT,
                            const float* __restrict__ bias,
                            _Float16* __restrict__ Y,
                            int Fin, int transpose_out) {
  const int tid  = threadIdx.x;
  const int lane = tid & 31;
  const int wv   = tid >> 5;
  const int l15  = lane & 15;
  const int hi   = lane >> 4;
  const int kbA  = hi * 8;
  const int kbB  = hi * 8;
  const int b = blockIdx.z, h = blockIdx.y;
  const int m0 = blockIdx.x * 128 + wv * 16;
  const size_t bh = (size_t)b * H_ + h;

  const float* Xrow = X + (bh * S_ + m0 + l15) * (size_t)Fin;   // this lane's A row
  const _Float16* Wh = WT + (size_t)h * D_ * Fin;

  v8f acc[8] = {};
  const int ksteps = Fin >> 5;
  for (int c = 0; c < ksteps; ++c) {
    if (c + 1 < ksteps)  // prefetch next A chunk (global_prefetch_b8)
      __builtin_prefetch(Xrow + 32 * (c + 1) + kbA, 0, 0);
    v16h a = load_a_f32(Xrow, 32 * c + kbA);
#pragma unroll
    for (int t = 0; t < 8; ++t) {
      v16h bf = load_b_frag(Wh + (size_t)(16 * t + l15) * Fin, 32 * c + kbB);
      acc[t] = wmma_fp16(a, bf, acc[t]);
    }
  }

  if (!transpose_out) {
#pragma unroll
    for (int t = 0; t < 8; ++t) {
      float bn = bias[h * D_ + 16 * t + l15];
#pragma unroll
      for (int r = 0; r < 8; ++r)
        Y[(bh * S_ + m0 + 8 * hi + r) * D_ + 16 * t + l15] =
            (_Float16)(acc[t][r] + bn);
    }
  } else {
    // transposed store: lane holds column d=16t+l15, rows s=m0+8*hi+r (contiguous)
#pragma unroll
    for (int t = 0; t < 8; ++t) {
      float bn = bias[h * D_ + 16 * t + l15];
      v8h pack;
#pragma unroll
      for (int r = 0; r < 8; ++r) pack[r] = (_Float16)(acc[t][r] + bn);
      *(v8h*)(Y + (bh * D_ + 16 * t + l15) * (size_t)S_ + m0 + 8 * hi) = pack;
    }
  }
}

// ------------------------------------- K2: flash attention (online softmax)
// Q,K f16 [B,H,S,128]; Vt f16 [B,H,128,S]; O f16 [B,S,DIM] (head-concat layout)
// K/V tiles double-buffered in LDS, filled by GLOBAL_LOAD_ASYNC_TO_LDS_B128.
__global__ void flash_kernel(const _Float16* __restrict__ Q,
                             const _Float16* __restrict__ K,
                             const _Float16* __restrict__ Vt,
                             _Float16* __restrict__ O) {
  __shared__ _Float16 Kt_lds[2][32][136];   // 32 keys x 128 d   (padded stride)
  __shared__ _Float16 Vt_lds[2][128][40];   // 128 d x 32 keys   (padded stride)
  __shared__ _Float16 P_lds[8][16][40];     // per-wave exp(P) 16x32 staging

  const int tid  = threadIdx.x;
  const int lane = tid & 31;
  const int wv   = tid >> 5;
  const int l15  = lane & 15;
  const int hi   = lane >> 4;
  const int kbA  = hi * 8;
  const int kbB  = hi * 8;
  const int b = blockIdx.z, h = blockIdx.y;
  const int q0 = blockIdx.x * 128 + wv * 16;
  const size_t bh = (size_t)b * H_ + h;

  // this thread's share of the cooperative K/V tile staging
  const int kr = tid >> 3, kc = (tid & 7) * 16;   // K tile: 32 rows x 128 cols
  const int vr = tid >> 1, vc = (tid & 1) * 16;   // V^T tile: 128 rows x 32 cols
  const _Float16* Kg = K  + (bh * S_ + kr) * (size_t)D_ + kc;  // advance by kv*D_
  const _Float16* Vg = Vt + (bh * D_ + vr) * (size_t)S_ + vc;  // advance by kv

  // issue async stage of tile 0 into buffer 0
  {
    async_ld_b128(&Kt_lds[0][kr][kc],     Kg);
    async_ld_b128(&Kt_lds[0][kr][kc + 8], Kg + 8);
    async_ld_b128(&Vt_lds[0][vr][vc],     Vg);
    async_ld_b128(&Vt_lds[0][vr][vc + 8], Vg + 8);
  }

  // Q fragments for this wave's 16 query rows (D=128 -> 4 k-chunks of 32);
  // these ordinary loads overlap the first async tile fill.
  const _Float16* Qrow = Q + (bh * S_ + q0 + l15) * (size_t)D_;
  v16h qf[4];
#pragma unroll
  for (int c = 0; c < 4; ++c) qf[c] = load_a_f16(Qrow, 32 * c + kbA);

  v8f acc[8] = {};                 // O accum: 8 d-tiles x 16x16
  float Mrow[8], Lrow[8];
#pragma unroll
  for (int r = 0; r < 8; ++r) { Mrow[r] = -1e30f; Lrow[r] = 0.0f; }

  wait_async0();
  __syncthreads();

  int cur = 0;
  for (int kv = 0; kv < S_; kv += 32) {
    // --- kick off async fill of the other buffer with the next tile
    if (kv + 32 < S_) {
      const int nxt = cur ^ 1;
      const _Float16* kn = Kg + (size_t)(kv + 32) * D_;
      const _Float16* vn = Vg + (kv + 32);
      async_ld_b128(&Kt_lds[nxt][kr][kc],     kn);
      async_ld_b128(&Kt_lds[nxt][kr][kc + 8], kn + 8);
      async_ld_b128(&Vt_lds[nxt][vr][vc],     vn);
      async_ld_b128(&Vt_lds[nxt][vr][vc + 8], vn + 8);
    }

    // --- scores: two 16x16 tiles over 32 keys, K-dim = 128 (4 WMMA each)
    v8f c0 = {}, c1 = {};
#pragma unroll
    for (int c = 0; c < 4; ++c) {
      v16h b0 = load_b_frag(&Kt_lds[cur][l15][0],      32 * c + kbB);
      v16h b1 = load_b_frag(&Kt_lds[cur][16 + l15][0], 32 * c + kbB);
      c0 = wmma_fp16(qf[c], b0, c0);
      c1 = wmma_fp16(qf[c], b1, c1);
    }

    // --- online softmax on C-fragment layout (row r+8*hi across 16 lanes)
#pragma unroll
    for (int r = 0; r < 8; ++r) {
      float s0 = c0[r] * SCALE_;
      float s1 = c1[r] * SCALE_;
      float rm = rowmax16(fmaxf(s0, s1));
      float mnew  = fmaxf(Mrow[r], rm);
      float alpha = __expf(Mrow[r] - mnew);
      Mrow[r] = mnew;
      float p0 = __expf(s0 - mnew);
      float p1 = __expf(s1 - mnew);
      Lrow[r] = Lrow[r] * alpha + rowsum16(p0 + p1);
      P_lds[wv][8 * hi + r][l15]      = (_Float16)p0;
      P_lds[wv][8 * hi + r][16 + l15] = (_Float16)p1;
#pragma unroll
      for (int t = 0; t < 8; ++t) acc[t][r] *= alpha;
    }
    __syncthreads();   // publish P (C-layout -> A-layout round trip through LDS)

    // --- O += P(16x32) * V(32x16) for each of 8 d-tiles
    v16h pf = load_a_f16(&P_lds[wv][l15][0], kbA);
#pragma unroll
    for (int t = 0; t < 8; ++t) {
      v16h vb = load_b_frag(&Vt_lds[cur][16 * t + l15][0], kbB);
      acc[t] = wmma_fp16(pf, vb, acc[t]);
    }

    wait_async0();     // next tile's async DMA (had all of compute to land)
    __syncthreads();   // all waves done reading cur + all async fills visible
    cur ^= 1;
  }

  // --- epilogue: normalize and store f16 in [B,S,H*D] (head-concat) layout
  float inv[8];
#pragma unroll
  for (int r = 0; r < 8; ++r) inv[r] = 1.0f / Lrow[r];
#pragma unroll
  for (int t = 0; t < 8; ++t) {
#pragma unroll
    for (int r = 0; r < 8; ++r) {
      O[((size_t)b * S_ + q0 + 8 * hi + r) * DIM_ + h * D_ + 16 * t + l15] =
          (_Float16)(acc[t][r] * inv[r]);
    }
  }
}

// ------------------------------------------ K3: output projection (WMMA)
// out f32 [B*S][DIM] = O f16 [B*S][DIM] @ Wo + bo;  WoT f16 [DIM][DIM] (= Wo^T)
// Each wave computes 16 rows x 128 cols (A-fragment reused 8x).
__global__ void outproj_kernel(const _Float16* __restrict__ O,
                               const _Float16* __restrict__ WoT,
                               const float* __restrict__ bo,
                               float* __restrict__ out) {
  const int tid  = threadIdx.x;
  const int lane = tid & 31;
  const int wv   = tid >> 5;
  const int l15  = lane & 15;
  const int hi   = lane >> 4;
  const int kbA  = hi * 8;
  const int kbB  = hi * 8;
  const int m0 = blockIdx.y * 128 + wv * 16;
  const int n0 = blockIdx.x * 128;

  const _Float16* Arow = O + (size_t)(m0 + l15) * DIM_;
  v8f acc[8] = {};
  for (int c = 0; c < DIM_ / 32; ++c) {
    if (c + 1 < DIM_ / 32)
      __builtin_prefetch(Arow + 32 * (c + 1) + kbA, 0, 0);
    v16h a = load_a_f16(Arow, 32 * c + kbA);
#pragma unroll
    for (int t = 0; t < 8; ++t) {
      v16h bf = load_b_frag(WoT + (size_t)(n0 + 16 * t + l15) * DIM_, 32 * c + kbB);
      acc[t] = wmma_fp16(a, bf, acc[t]);
    }
  }
#pragma unroll
  for (int t = 0; t < 8; ++t) {
    int col = n0 + 16 * t + l15;
    float bn = bo[col];
#pragma unroll
    for (int r = 0; r < 8; ++r)
      out[(size_t)(m0 + 8 * hi + r) * DIM_ + col] = acc[t][r] + bn;
  }
}

// --------------------------------------------------------------- launcher
extern "C" void kernel_launch(void* const* d_in, const int* in_sizes, int n_in,
                              void* d_out, int out_size, void* d_ws, size_t ws_size,
                              hipStream_t stream) {
  (void)in_sizes; (void)n_in; (void)out_size; (void)ws_size;
  const float* q  = (const float*)d_in[0];
  const float* k  = (const float*)d_in[1];
  const float* v  = (const float*)d_in[2];
  const float* Wq = (const float*)d_in[3];
  const float* Wk = (const float*)d_in[4];
  const float* Wv = (const float*)d_in[5];
  const float* bq = (const float*)d_in[6];
  const float* bk = (const float*)d_in[7];
  const float* bv = (const float*)d_in[8];
  const float* Wo = (const float*)d_in[9];
  const float* bo = (const float*)d_in[10];
  float* out = (float*)d_out;

  char* ws = (char*)d_ws;
  size_t off = 0;
  auto carve = [&](size_t halves) -> _Float16* {
    _Float16* p = (_Float16*)(ws + off);
    off += ((halves * sizeof(_Float16) + 255) / 256) * 256;
    return p;
  };
  _Float16* WqT = carve((size_t)H_ * F_ * D_);      // [H][128][256]
  _Float16* WkT = carve((size_t)H_ * F_ * D_);      // [H][128][256]
  _Float16* WvT = carve((size_t)H_ * D_ * D_);      // [H][128][128]
  _Float16* WoT = carve((size_t)DIM_ * DIM_);       // [2048][2048]
  _Float16* Qf  = carve((size_t)B_ * H_ * S_ * D_); // [B,H,S,128]
  _Float16* Kf  = carve((size_t)B_ * H_ * S_ * D_); // [B,H,S,128]
  _Float16* Vtf = carve((size_t)B_ * H_ * D_ * S_); // [B,H,128,S]
  _Float16* Of  = carve((size_t)B_ * S_ * DIM_);    // [B,S,2048]

  // K0: repack weights (f32 -> f16, transposed)
  cvt_transpose_kernel<<<2048, 256, 0, stream>>>(Wq, WqT, F_, D_, H_);
  cvt_transpose_kernel<<<2048, 256, 0, stream>>>(Wk, WkT, F_, D_, H_);
  cvt_transpose_kernel<<<1024, 256, 0, stream>>>(Wv, WvT, D_, D_, H_);
  cvt_transpose_kernel<<<4096, 256, 0, stream>>>(Wo, WoT, DIM_, DIM_, 1);

  // K1: projections
  dim3 pg(S_ / 128, H_, B_);
  proj_kernel<<<pg, 256, 0, stream>>>(q, WqT, bq, Qf,  F_, 0);
  proj_kernel<<<pg, 256, 0, stream>>>(k, WkT, bk, Kf,  F_, 0);
  proj_kernel<<<pg, 256, 0, stream>>>(v, WvT, bv, Vtf, D_, 1);

  // K2: flash attention (async double-buffered K/V staging)
  flash_kernel<<<pg, 256, 0, stream>>>(Qf, Kf, Vtf, Of);

  // K3: output projection
  outproj_kernel<<<dim3(DIM_ / 128, (B_ * S_) / 128), 256, 0, stream>>>(Of, WoT, bo, out);
}
